// CINLayer_82935818486506
// MI455X (gfx1250) — matrix operand order repaired and checked by reading, not compiled, not run on hardware
//
#include <hip/hip_runtime.h>
#include <stdint.h>

// CIN (xDeepFM) on MI455X / gfx1250, bf16 WMMA path + TDM staging.
//
// Reference: Xn[b,h,d] = sum_{j,k} W[h, j*Fi+k] * X0[b,j,d] * Xi[b,k,d] + bias[h]
// Restructured per d as:  acc[h,n] = bias[h] + sum_j s_j(n) * (W_j [128,Fi] @ Xi_d [Fi,128])
// where s_j(n) = X0[b_n, j, d] is folded in with per-column v_pk_fma on the WMMA output.
//
// Workspace map (requires ws_size >= 36 MiB):
//   [0, ~3.05MB)       W in bf16, pre-swizzled to WMMA A-operand lane layout
//   [4MiB, 20MiB)      X ping buffer, bf16, layout [d][b][128]
//   [20MiB, 36MiB)     X pong buffer, bf16, layout [d][b][128]

#ifndef USE_TDM
#define USE_TDM 1
#endif

typedef __attribute__((ext_vector_type(16))) __bf16 v16bf;
typedef __attribute__((ext_vector_type(8)))  float  v8f;
typedef __attribute__((ext_vector_type(4)))  unsigned int u32x4;
typedef __attribute__((ext_vector_type(8)))  int          i32x8;
typedef __attribute__((ext_vector_type(4)))  int          i32x4;

#define B_TOT   4096
#define F0N     39
#define D_EMB   16
#define H_OUT   128
#define NTILE   128   // batch columns per workgroup
#define NSUB    8     // 8 x 16-column WMMA subtiles

__device__ __forceinline__ unsigned short f2bf(float f) {
  unsigned int u = __float_as_uint(f);
  u += 0x7FFFu + ((u >> 16) & 1u);     // round-to-nearest-even
  return (unsigned short)(u >> 16);
}
__device__ __forceinline__ float bf2f(unsigned short s) {
  return __uint_as_float(((unsigned int)s) << 16);
}

// ---------------------------------------------------------------------------
// Convert W (f32 [128][F0*Fi]) -> bf16 tiles in a WMMA A-operand lane order (row per lane, contiguous K per half).
// Tile id = (ht*F0 + j)*KC + kc ; each tile is 32 lanes x 32 bytes = 1 KB.
// Lane L holds row h = 16*ht + L%16 ; K = (L/16)*16 + 0..15 (contiguous).
// ---------------------------------------------------------------------------
__global__ void prep_w(const float* __restrict__ W, unsigned short* __restrict__ Wws,
                       int Fi, int KC, int Cin) {
  int gidx = blockIdx.x * blockDim.x + threadIdx.x;
  int ntiles = 8 * F0N * KC;
  if (gidx >= ntiles * 32) return;
  int tile = gidx >> 5, lane = gidx & 31;
  int ht = tile / (F0N * KC);
  int j  = (tile / KC) % F0N;
  int kc = tile % KC;
  int h    = ht * 16 + (lane & 15);
  int half = lane >> 4;

  unsigned short o[16];
#pragma unroll
  for (int c = 0; c < 2; ++c) {
    int kbase = kc * 32 + half * 16 + c * 8;
#pragma unroll
    for (int k = 0; k < 8; ++k) {
      int kk = kbase + k;
      float v = (kk < Fi) ? W[(size_t)h * Cin + (size_t)j * Fi + kk] : 0.0f;
      o[c * 8 + k] = f2bf(v);
    }
  }
  unsigned short* dst = Wws + (size_t)tile * 512 + (size_t)lane * 16;
  ((uint4*)dst)[0] = ((const uint4*)o)[0];
  ((uint4*)dst)[1] = ((const uint4*)o)[1];
}

// ---------------------------------------------------------------------------
// X0 (f32 [B][39][16]) -> bf16 buffer [d][b][128] with rows 39..63 zeroed
// (zero padding makes the K-padded layer-1 WMMA chunks exact).
// ---------------------------------------------------------------------------
__global__ void prep_x(const float* __restrict__ X0, unsigned short* __restrict__ Xb) {
  int idx = blockIdx.x * blockDim.x + threadIdx.x;   // [0, 4096*64)
  int b = idx >> 6, slot = idx & 63;
  if (b >= B_TOT) return;
  if (slot < F0N) {
    const float* src = X0 + ((size_t)b * F0N + slot) * D_EMB;
#pragma unroll
    for (int d = 0; d < D_EMB; ++d)
      Xb[((size_t)d * B_TOT + b) * H_OUT + slot] = f2bf(src[d]);
  } else {
#pragma unroll
    for (int d = 0; d < D_EMB; ++d)
      Xb[((size_t)d * B_TOT + b) * H_OUT + slot] = 0;
  }
}

// ---------------------------------------------------------------------------
// Main CIN layer. Grid: (B/128, 16=d). Block: 256 threads = 8 waves.
// Wave w owns output rows h = 16w..16w+15 for 128 batch columns, one d.
// ---------------------------------------------------------------------------
__global__ __launch_bounds__(256) void cin_layer(
    const unsigned short* __restrict__ Wws,   // bf16 swizzled tiles
    const float* __restrict__ X0raw,          // original f32 X0 (for per-column scales)
    const float* __restrict__ bias,
    const unsigned short* __restrict__ Xin,   // bf16 [d][b][128]
    unsigned short* __restrict__ Xout,        // bf16 [d][b][128]
    int Fi, int KC) {
  __shared__ __align__(32) unsigned short Bst[4 * 2 * NTILE * 16]; // 32 KB max
  __shared__ float X0s[NTILE * 40];                                 // padded stride 40
  __shared__ float biass[H_OUT];

  const int tid  = threadIdx.x;
  const int lane = tid & 31;
  const int wv   = tid >> 5;
  const int ln16 = lane & 15;
  const int g    = lane >> 4;      // K-group select for B operand / row-half for C
  const int bt   = blockIdx.x * NTILE;
  const int d    = blockIdx.y;

  // ---- Stage X0 scalars (f32) for the per-column FMA scales ----------------
  for (int idx = tid; idx < NTILE * F0N; idx += 256) {
    int n = idx / F0N, j = idx - n * F0N;
    X0s[n * 40 + j] = X0raw[((size_t)(bt + n) * F0N + j) * D_EMB + d];
  }
  if (tid < H_OUT) biass[tid] = bias[tid];

  // ---- Stage Xi tile into LDS in per-lane WMMA B-operand layout (contiguous K, as A) ----------
  // Entry e = (kc*2+g)*128 + n holds Xi[bt+n, kc*32+g*16 .. +16) for column n
  // as 16 contiguous bf16 (32 B), so the hot-loop B fetch is 2x ds_load_b128.
  // LDS is written sequentially in entry order; memory side is an affine 3-D
  // walk: dim0 = 16 contiguous elems, dim1 = n (stride 128 el), dim2 = q
  // (stride 16 el). That is exactly one TENSOR_LOAD_TO_LDS descriptor.
#if USE_TDM && defined(__gfx1250__) && \
    __has_builtin(__builtin_amdgcn_tensor_load_to_lds)
  if (wv == 0) {
    unsigned long long gaddr =
        (unsigned long long)(const void*)(Xin + ((size_t)(d * B_TOT + bt)) * H_OUT);
    unsigned int ldsoff = (unsigned int)(uintptr_t)(void*)&Bst[0];

    u32x4 g0;
    g0[0] = 1u;                                        // count=1, user mode
    g0[1] = ldsoff;                                    // lds_addr (bytes)
    g0[2] = (unsigned int)gaddr;                       // global_addr[31:0]
    g0[3] = (unsigned int)(gaddr >> 32) | (2u << 30);  // addr[56:32] | type=2

    i32x8 g1;
    g1[0] = 0x10000;                   // data_size=1 (2 bytes/elem)
    g1[1] = (int)(128u << 16);         // tensor_dim0[15:0] = 128
    g1[2] = (int)(128u << 16);         // tensor_dim0 hi=0 | tensor_dim1[15:0]=128
    g1[3] = (int)(16u << 16);          // tensor_dim1 hi=0 | tile_dim0=16
    g1[4] = (int)(128u | ((unsigned)(2 * KC) << 16)); // tile_dim1=128, tile_dim2=2*KC
    g1[5] = 128;                       // tensor_dim0_stride = 128 elems
    g1[6] = (int)(16u << 16);          // dim0_stride hi=0 | tensor_dim1_stride=16
    g1[7] = 0;                         // tensor_dim1_stride hi

    i32x4 g2;
    g2[0] = 8;                         // tensor_dim2 (z < 8)
    g2[1] = 1;                         // tensor_dim3 (unused)
    g2[2] = 0;                         // tensor_dim2_stride lo
    g2[3] = 0;                         // dim2_stride hi | tile_dim3=0
    i32x4 g3 = {0, 0, 0, 0};           // dims 3/4 unused
    i32x8 g4 = {0, 0, 0, 0, 0, 0, 0, 0}; // extra words in 6-arg form: zero-filled

    // 6-arg form on this toolchain: (u32x4, i32x8, i32x4, i32x4, i32x8, i32 cpol)
    __builtin_amdgcn_tensor_load_to_lds(g0, g1, g2, g3, g4, 0);
    __builtin_amdgcn_s_wait_tensorcnt(0);
  }
  // DMA writes to LDS are invisible to alias analysis; keep the compiler from
  // reasoning about Bst's contents.
  asm volatile("" ::: "memory");
#else
  const int NE = KC * 2 * NTILE;
  for (int e = tid; e < NE; e += 256) {
    int q = e >> 7, n = e & (NTILE - 1);
    const uint4* s4 =
        (const uint4*)(Xin + ((size_t)(d * B_TOT + bt + n)) * H_OUT + q * 16);
    uint4* d4 = (uint4*)(Bst + (size_t)e * 16);
    d4[0] = s4[0];
    d4[1] = s4[1];
  }
#endif
  __syncthreads();

  const int mrow = g * 8;  // first of this lane's 8 accumulator rows
  float binit[8];
#pragma unroll
  for (int i = 0; i < 8; ++i) binit[i] = biass[wv * 16 + mrow + i];

  v8f acc[NSUB];
#pragma unroll
  for (int ns = 0; ns < NSUB; ++ns)
#pragma unroll
    for (int i = 0; i < 8; ++i) acc[ns][i] = binit[i];

  v8f zeroc = {};

  for (int kc = 0; kc < KC; ++kc) {
    // B operands for this K-chunk live in registers across all 39 j's.
    v16bf Breg[NSUB];
#pragma unroll
    for (int ns = 0; ns < NSUB; ++ns) {
      int entry = (kc * 2 + g) * NTILE + ns * 16 + ln16;
      Breg[ns] = *(const v16bf*)(Bst + (size_t)entry * 16);
    }
    for (int j = 0; j < F0N; ++j) {
      const v16bf Av = *(const v16bf*)(
          Wws + ((size_t)((wv * F0N + j) * KC + kc)) * 512 + (size_t)lane * 16);
      float sc[NSUB];
#pragma unroll
      for (int ns = 0; ns < NSUB; ++ns)
        sc[ns] = X0s[(ns * 16 + ln16) * 40 + j];
#pragma unroll
      for (int ns = 0; ns < NSUB; ++ns) {
        v8f t = __builtin_amdgcn_wmma_f32_16x16x32_bf16(
            false, Av, false, Breg[ns], (short)0, zeroc, false, false);
#pragma unroll
        for (int i = 0; i < 8; ++i)
          acc[ns][i] = fmaf(sc[ns], t[i], acc[ns][i]);
      }
    }
  }

  // ---- Store Xn as bf16 [d][b][128]; each lane packs 8 row-contiguous vals --
#pragma unroll
  for (int ns = 0; ns < NSUB; ++ns) {
    unsigned short o[8];
#pragma unroll
    for (int i = 0; i < 8; ++i) o[i] = f2bf(acc[ns][i]);
    size_t off = ((size_t)(d * B_TOT + bt + ns * 16 + ln16)) * H_OUT + wv * 16 + mrow;
    *(uint4*)(Xout + off) = *(const uint4*)o;
  }
}

// ---------------------------------------------------------------------------
// Sum over d: out[b, off + h] = sum_d X[d][b][h]
// ---------------------------------------------------------------------------
__global__ void reduce_d(const unsigned short* __restrict__ X,
                         float* __restrict__ out, int off) {
  int idx = blockIdx.x * blockDim.x + threadIdx.x;   // [0, 4096*128)
  int b = idx >> 7, h = idx & 127;
  if (b >= B_TOT) return;
  float s = 0.0f;
#pragma unroll
  for (int d = 0; d < D_EMB; ++d)
    s += bf2f(X[((size_t)d * B_TOT + b) * H_OUT + h]);
  out[(size_t)b * 384 + off + h] = s;
}

// ---------------------------------------------------------------------------
extern "C" void kernel_launch(void* const* d_in, const int* in_sizes, int n_in,
                              void* d_out, int out_size, void* d_ws, size_t ws_size,
                              hipStream_t stream) {
  const float* X0 = (const float*)d_in[0];
  const float* W0 = (const float*)d_in[1];
  const float* b0 = (const float*)d_in[2];
  const float* W1 = (const float*)d_in[3];
  const float* b1 = (const float*)d_in[4];
  const float* W2 = (const float*)d_in[5];
  const float* b2 = (const float*)d_in[6];
  float* out = (float*)d_out;

  // workspace layout
  unsigned short* Wws = (unsigned short*)d_ws;
  const size_t WT0 = (size_t)8 * F0N * 2;   // 624 tiles (layer 1, Fi=39 padded to 64)
  const size_t WT1 = (size_t)8 * F0N * 4;   // 1248 tiles (layers 2/3, Fi=128)
  unsigned short* Wb0 = Wws;
  unsigned short* Wb1 = Wb0 + WT0 * 512;
  unsigned short* Wb2 = Wb1 + WT1 * 512;
  unsigned short* Xb0 = (unsigned short*)((char*)d_ws + (size_t)(4u << 20));
  unsigned short* Xb1 = Xb0 + (size_t)D_EMB * B_TOT * H_OUT;

  prep_w<<<(int)((WT0 * 32 + 255) / 256), 256, 0, stream>>>(W0, Wb0, 39, 2, 39 * 39);
  prep_w<<<(int)((WT1 * 32 + 255) / 256), 256, 0, stream>>>(W1, Wb1, 128, 4, 39 * 128);
  prep_w<<<(int)((WT1 * 32 + 255) / 256), 256, 0, stream>>>(W2, Wb2, 128, 4, 39 * 128);
  prep_x<<<(B_TOT * 64) / 256, 256, 0, stream>>>(X0, Xb0);

  dim3 grid(B_TOT / NTILE, D_EMB);   // (32, 16)
  cin_layer<<<grid, 256, 0, stream>>>(Wb0, X0, b0, Xb0, Xb1, 39, 2);
  reduce_d<<<(B_TOT * H_OUT) / 256, 256, 0, stream>>>(Xb1, out, 0);
  cin_layer<<<grid, 256, 0, stream>>>(Wb1, X0, b1, Xb1, Xb0, 128, 4);
  reduce_d<<<(B_TOT * H_OUT) / 256, 256, 0, stream>>>(Xb0, out, 128);
  cin_layer<<<grid, 256, 0, stream>>>(Wb2, X0, b2, Xb0, Xb1, 128, 4);
  reduce_d<<<(B_TOT * H_OUT) / 256, 256, 0, stream>>>(Xb1, out, 256);
}
